// Model_58626303590511
// MI455X (gfx1250) — matrix operand order repaired and checked by reading, not compiled, not run on hardware
//
#include <hip/hip_runtime.h>

typedef __bf16 bf16;
typedef bf16 v16bf __attribute__((ext_vector_type(16)));
typedef bf16 v8bf  __attribute__((ext_vector_type(8)));
typedef float v8f  __attribute__((ext_vector_type(8)));

#define PI_D 3.14159265358979323846
#define PI_F 3.14159265358979323846f
#define EPSF 1e-5f

__device__ __forceinline__ int iabs_(int x){ return x < 0 ? -x : x; }
__device__ __forceinline__ int imin_(int a,int b){ return a < b ? a : b; }
__device__ __forceinline__ int imax_(int a,int b){ return a > b ? a : b; }

// ---------------------------------------------------------------------------
// Wigner-d and Driscoll-Healy quadrature (double precision, computed on device)
// ---------------------------------------------------------------------------
__device__ double lfact_(int n){ return lgamma((double)n + 1.0); }
__device__ double ipow_(double x, int n){ double r = 1.0; for (int i = 0; i < n; ++i) r *= x; return r; }

// d^l_{mrow,mcol}(beta) = <l mrow | e^{-i beta Jy} | l mcol>
__device__ double wigner_d(int l, int mrow, int mcol, double beta){
  int mp = mrow, m = mcol;
  if (iabs_(mp) > l || iabs_(m) > l) return 0.0;
  double cb = cos(0.5 * beta), sb = sin(0.5 * beta);
  int smin = imax_(0, m - mp);
  int smax = imin_(l + m, l - mp);
  if (smax < smin) return 0.0;
  double pref = 0.5 * (lfact_(l+mp) + lfact_(l-mp) + lfact_(l+m) + lfact_(l-m));
  double acc = 0.0;
  for (int s = smin; s <= smax; ++s){
    double ln = pref - lfact_(l+m-s) - lfact_(s) - lfact_(mp-m+s) - lfact_(l-mp-s);
    double t  = exp(ln) * ipow_(cb, 2*l + m - mp - 2*s) * ipow_(sb, mp - m + 2*s);
    acc += ((mp - m + s) & 1) ? -t : t;
  }
  return acc;
}

__device__ double quadw_(int bg, int k){
  double beta = PI_D * (2*k + 1) / (4.0 * bg);
  double s = 0.0;
  for (int j = 0; j < bg; ++j) s += sin(beta * (2*j + 1)) / (double)(2*j + 1);
  return (2.0 / bg) * sin(beta) * s;
}

// ---------------------------------------------------------------------------
// Constant-fill kernels
// ---------------------------------------------------------------------------
__global__ void k_fill_s2_ana(float* out, int L, int K, int J, int bg, float scale){
  int idx = blockIdx.x * blockDim.x + threadIdx.x;
  int total = L * K * J; if (idx >= total) return;
  int j = idx % J, k = (idx / J) % K, l = idx / (J * K);
  int m = j - (L - 1);
  float v = 0.f;
  if (iabs_(m) <= l){
    double beta = PI_D * (2*k + 1) / (4.0 * bg);
    v = (float)(wigner_d(l, m, 0, beta) * quadw_(bg, k) * (double)scale);
  }
  out[idx] = v;
}

__global__ void k_fill_half1(float* out, int L, int J){
  int idx = blockIdx.x * blockDim.x + threadIdx.x;
  if (idx >= L * J) return;
  int j = idx % J, l = idx / J;
  int m = j - (L - 1);
  out[idx] = (iabs_(m) <= l) ? (float)wigner_d(l, m, 0, 0.5 * PI_D) : 0.f;
}

__global__ void k_fill_half2d(float* out, int L, int J){
  int idx = blockIdx.x * blockDim.x + threadIdx.x;
  int total = L * J * J; if (idx >= total) return;
  int n = idx % J, j = (idx / J) % J, l = idx / (J * J);
  int mj = j - (L - 1), mn = n - (L - 1);
  out[idx] = (iabs_(mj) <= l && iabs_(mn) <= l)
           ? (float)wigner_d(l, mj, mn, 0.5 * PI_D) : 0.f;
}

// isSyn=1: d * (2l+1);  isSyn=0: d * quadw(bg,k) * scale
__global__ void k_fill_wig4(float* out, int L, int K, int J, int bg, int isSyn, float scale){
  int idx = blockIdx.x * blockDim.x + threadIdx.x;
  int total = L * K * J * J; if (idx >= total) return;
  int n = idx % J, j = (idx / J) % J, k = (idx / (J * J)) % K, l = idx / (J * J * K);
  int mj = j - (L - 1), mn = n - (L - 1);
  float v = 0.f;
  if (iabs_(mj) <= l && iabs_(mn) <= l){
    double beta = PI_D * (2*k + 1) / (4.0 * bg);
    double d = wigner_d(l, mj, mn, beta);
    v = isSyn ? (float)(d * (2*l + 1)) : (float)(d * quadw_(bg, k) * (double)scale);
  }
  out[idx] = v;
}

// ---------------------------------------------------------------------------
// Forward DFT kernels
// ---------------------------------------------------------------------------
// Layer1: xh[(b*F+f)*K+k][j] = sum_a x[...][a] e^{-2pi i m_j a / Nin}
__global__ void k_dft_alpha(const float* __restrict__ x, float2* __restrict__ xh,
                            int Rows, int Nin, int J, int Lm1){
  int idx = blockIdx.x * blockDim.x + threadIdx.x;
  int total = Rows * J; if (idx >= total) return;
  int j = idx % J, r = idx / J;
  int m = j - Lm1;
  const float* row = x + (size_t)r * Nin;
  float w = -2.f * PI_F * (float)m / (float)Nin;
  float re = 0.f, im = 0.f;
  for (int a = 0; a < Nin; ++a){
    float s, c; sincosf(w * a, &s, &c);
    float v = row[a]; re += v * c; im += v * s;
  }
  xh[idx] = make_float2(re, im);
}

// Layers 2/3: xh[g][j][n] = sum_{u,v} x[g][u][v] e^{+2pi i (m_j u + m_n v)/Nin}
__global__ void k_dft2(const float* __restrict__ x, float2* __restrict__ xh,
                       int Nin, int J, int Lm1){
  extern __shared__ char smemraw[];
  float*  xs = (float*)smemraw;                 // Nin*Nin
  float2* H  = (float2*)(xs + Nin * Nin);       // J*Nin
  int g = blockIdx.x;
  const float* src = x + (size_t)g * Nin * Nin;
  for (int i = threadIdx.x; i < Nin * Nin; i += blockDim.x) xs[i] = src[i];
  __syncthreads();
  for (int t = threadIdx.x; t < J * Nin; t += blockDim.x){
    int j = t / Nin, v = t % Nin;
    float w = 2.f * PI_F * (float)(j - Lm1) / (float)Nin;
    float re = 0.f, im = 0.f;
    for (int u = 0; u < Nin; ++u){
      float s, c; sincosf(w * u, &s, &c);
      float val = xs[u * Nin + v]; re += val * c; im += val * s;
    }
    H[t] = make_float2(re, im);
  }
  __syncthreads();
  float2* dst = xh + (size_t)g * J * J;
  for (int t = threadIdx.x; t < J * J; t += blockDim.x){
    int j = t / J, n = t % J;
    float w = 2.f * PI_F * (float)(n - Lm1) / (float)Nin;
    float re = 0.f, im = 0.f;
    const float2* Hj = H + j * Nin;
    for (int v = 0; v < Nin; ++v){
      float s, c; sincosf(w * v, &s, &c);
      float2 h = Hj[v];
      re += h.x * c - h.y * s;
      im += h.x * s + h.y * c;
    }
    dst[t] = make_float2(re, im);
  }
}

// ---------------------------------------------------------------------------
// Analysis projections (Fx) and kernel spectra (Ky)
// ---------------------------------------------------------------------------
// Fx[l][j][b][f] = sum_k ana[l][k][j] * xh[(b*F+f)*K+k][j]
__global__ void k_fx_s2(const float* __restrict__ ana, const float2* __restrict__ xh,
                        float2* __restrict__ Fx, int L, int K, int J, int B, int F){
  long idx = (long)blockIdx.x * blockDim.x + threadIdx.x;
  long total = (long)L * J * B * F; if (idx >= total) return;
  int f = idx % F, b = (idx / F) % B, j = (idx / ((long)F * B)) % J, l = idx / ((long)F * B * J);
  float re = 0.f, im = 0.f;
  for (int k = 0; k < K; ++k){
    float a = ana[((size_t)l * K + k) * J + j];
    if (a != 0.f){
      float2 v = xh[(((size_t)b * F + f) * K + k) * J + j];
      re += a * v.x; im += a * v.y;
    }
  }
  Fx[idx] = make_float2(re, im);
}

// Fx[l][j][kap][b][f] = sum_k ana[l][k][j][kap] * xh[(b*F+f)*K+k][j][kap]
__global__ void k_fx_so3(const float* __restrict__ ana, const float2* __restrict__ xh,
                         float2* __restrict__ Fx, int L, int K, int J, int B, int F){
  long idx = (long)blockIdx.x * blockDim.x + threadIdx.x;
  long total = (long)L * J * J * B * F; if (idx >= total) return;
  int f   = idx % F;
  int b   = (idx / F) % B;
  int kap = (idx / ((long)F * B)) % J;
  int j   = (idx / ((long)F * B * J)) % J;
  int l   =  idx / ((long)F * B * J * J);
  float re = 0.f, im = 0.f;
  for (int k = 0; k < K; ++k){
    float a = ana[(((size_t)l * K + k) * J + j) * J + kap];
    if (a != 0.f){
      float2 v = xh[((((size_t)b * F + f) * K + k) * J + j) * J + kap];
      re += a * v.x; im += a * v.y;
    }
  }
  Fx[idx] = make_float2(re, im);
}

// Ky[l][j][f][o] = half1[l][j]*scale * sum_p e^{-i m_j 2pi p/P} w[f][o][p]
__global__ void k_ky_s2(const float* __restrict__ half1, const float* __restrict__ w,
                        float2* __restrict__ Ky, int L, int J, int F, int O, int P,
                        float scale){
  long idx = (long)blockIdx.x * blockDim.x + threadIdx.x;
  long total = (long)L * J * F * O; if (idx >= total) return;
  int o = idx % O, f = (idx / O) % F, j = (idx / ((long)O * F)) % J, l = idx / ((long)O * F * J);
  float h = half1[(size_t)l * J + j] * scale;
  float re = 0.f, im = 0.f;
  if (h != 0.f){
    int m = j - (L - 1);
    float wf = -2.f * PI_F * (float)m / (float)P;
    const float* wp = w + ((size_t)f * O + o) * P;
    for (int p = 0; p < P; ++p){
      float s, c; sincosf(wf * p, &s, &c);
      float v = wp[p]; re += v * c; im += v * s;
    }
    re *= h; im *= h;
  }
  Ky[idx] = make_float2(re, im);
}

// Ky[l][j][kap][f][o] = half[l][j][kap]*scale * sum_p e^{+i m_j 2pi p/P} w[f][o][p]
__global__ void k_ky_so3(const float* __restrict__ half, const float* __restrict__ w,
                         float2* __restrict__ Ky, int L, int J, int F, int O, int P,
                         float scale){
  long idx = (long)blockIdx.x * blockDim.x + threadIdx.x;
  long total = (long)L * J * J * F * O; if (idx >= total) return;
  int o   = idx % O;
  int f   = (idx / O) % F;
  int kap = (idx / ((long)O * F)) % J;
  int j   = (idx / ((long)O * F * J)) % J;
  int l   =  idx / ((long)O * F * J * J);
  float h = half[((size_t)l * J + j) * J + kap] * scale;
  float re = 0.f, im = 0.f;
  if (h != 0.f){
    int m = j - (L - 1);
    float wf = 2.f * PI_F * (float)m / (float)P;
    const float* wp = w + ((size_t)f * O + o) * P;
    for (int p = 0; p < P; ++p){
      float s, c; sincosf(wf * p, &s, &c);
      float v = wp[p]; re += v * c; im += v * s;
    }
    re *= h; im *= h;
  }
  Ky[idx] = make_float2(re, im);
}

// ---------------------------------------------------------------------------
// bf16 packing for the WMMA complex GEMM
// A rows = (j*B + b), cols = (kap*F + f).  Zero-padded to [Mp][Kp].
// ---------------------------------------------------------------------------
__global__ void k_packA(const float2* __restrict__ Fx, bf16* __restrict__ Ar, bf16* __restrict__ Ai,
                        int L, int J, int Kdim, int B, int F, int Mp, int Kp){
  long idx = (long)blockIdx.x * blockDim.x + threadIdx.x;
  long total = (long)L * Mp * Kp; if (idx >= total) return;
  int kk = idx % Kp, row = (idx / Kp) % Mp, l = idx / ((long)Kp * Mp);
  float re = 0.f, im = 0.f;
  if (row < J * B && kk < Kdim * F){
    int j = row / B, b = row % B, kap = kk / F, f = kk % F;
    float2 v = Fx[((((size_t)l * J + j) * Kdim + kap) * B + b) * F + f];
    re = v.x; im = v.y;
  }
  Ar[idx] = (bf16)re; Ai[idx] = (bf16)im;
}

// B rows = (n*O + o), cols = (kap*F + f). Packs P=Re(Ky), Q=Im(Ky), R=-Im(Ky).
__global__ void k_packB(const float2* __restrict__ Ky, bf16* __restrict__ P, bf16* __restrict__ Q,
                        bf16* __restrict__ R, int L, int J, int Kdim, int F, int O, int Np, int Kp){
  long idx = (long)blockIdx.x * blockDim.x + threadIdx.x;
  long total = (long)L * Np * Kp; if (idx >= total) return;
  int kk = idx % Kp, row = (idx / Kp) % Np, l = idx / ((long)Kp * Np);
  float re = 0.f, im = 0.f;
  if (row < J * O && kk < Kdim * F){
    int n = row / O, o = row % O, kap = kk / F, f = kk % F;
    float2 v = Ky[((((size_t)l * J + n) * Kdim + kap) * F + f) * O + o];
    re = v.x; im = v.y;
  }
  P[idx] = (bf16)re; Q[idx] = (bf16)im; R[idx] = (bf16)(-im);
}

// ---------------------------------------------------------------------------
// Batched complex GEMM:  Z_l = A_l * conj(Ky_l)^T  via v_wmma_f32_16x16x32_bf16
//   Zr = Ar*P + Ai*Q ; Zi = Ar*R + Ai*P   (P=Re, Q=Im, R=-Im)
// One wave (32 lanes) computes a 16x16 tile of Zr and Zi.
// ---------------------------------------------------------------------------
__global__ __launch_bounds__(32)
void k_zgemm(const bf16* __restrict__ Ar, const bf16* __restrict__ Ai,
             const bf16* __restrict__ Bp, const bf16* __restrict__ Bq, const bf16* __restrict__ Br,
             float* __restrict__ Zr, float* __restrict__ Zi, int Mp, int Np, int Kp){
  const int l    = blockIdx.z;
  const int m0   = blockIdx.x * 16;
  const int n0   = blockIdx.y * 16;
  const int lane = threadIdx.x;
  // A 16x32 bf16, k order used here: lanes<16: M=lane, K in 0..15; lanes>=16: K in 16..31
  const size_t aoff = (size_t)l * Mp * Kp + (size_t)(m0 + (lane & 15)) * Kp + ((lane & 16) ? 16 : 0);
  // B 32x16 bf16, same k order as A: lane holds column n=lane%16, K half = 0..15 (lanes<16) or 16..31
  const size_t boff = (size_t)l * Np * Kp + (size_t)(n0 + (lane & 15)) * Kp + ((lane & 16) ? 16 : 0);
  const bf16* par = Ar + aoff;
  const bf16* pai = Ai + aoff;
  const bf16* pbp = Bp + boff;
  const bf16* pbq = Bq + boff;
  const bf16* pbr = Br + boff;
  v8f cr = {}; v8f ci = {};
  for (int k0 = 0; k0 < Kp; k0 += 32){
    v16bf ar, ai;
    {
      v8bf lo = *reinterpret_cast<const v8bf*>(par + k0);
      v8bf hi = *reinterpret_cast<const v8bf*>(par + k0 + 8);
      #pragma unroll
      for (int e = 0; e < 8; ++e){ ar[e] = lo[e]; ar[8 + e] = hi[e]; }
      lo = *reinterpret_cast<const v8bf*>(pai + k0);
      hi = *reinterpret_cast<const v8bf*>(pai + k0 + 8);
      #pragma unroll
      for (int e = 0; e < 8; ++e){ ai[e] = lo[e]; ai[8 + e] = hi[e]; }
    }
    v16bf bp = *reinterpret_cast<const v16bf*>(pbp + k0);
    v16bf bq = *reinterpret_cast<const v16bf*>(pbq + k0);
    v16bf br = *reinterpret_cast<const v16bf*>(pbr + k0);
    cr = __builtin_amdgcn_wmma_f32_16x16x32_bf16(false, ar, false, bp, (short)0, cr, false, false);
    cr = __builtin_amdgcn_wmma_f32_16x16x32_bf16(false, ai, false, bq, (short)0, cr, false, false);
    ci = __builtin_amdgcn_wmma_f32_16x16x32_bf16(false, ar, false, br, (short)0, ci, false, false);
    ci = __builtin_amdgcn_wmma_f32_16x16x32_bf16(false, ai, false, bp, (short)0, ci, false, false);
  }
  // C f32 layout: VGPR r -> (M=r, N=lane) for lanes<16 ; (M=8+r, N=lane-16) otherwise
  const int mrow = (lane & 16) ? 8 : 0;
  const int nn   = n0 + (lane & 15);
  float* zr = Zr + (size_t)l * Mp * Np;
  float* zi = Zi + (size_t)l * Mp * Np;
  #pragma unroll
  for (int r = 0; r < 8; ++r){
    size_t o = (size_t)(m0 + mrow + r) * Np + nn;
    zr[o] = cr[r];
    zi[o] = ci[r];
  }
}

// ---------------------------------------------------------------------------
// Fused Wigner synthesis + inverse 2-D DFT (+bias). One workgroup per (b,o,k).
// y[b][o][k][u][v] = bias[o] + Re sum_{j,n} (sum_l syn[l][k][j][n] Z_l[j*B+b][n*O+o])
//                                * e^{-2pi i (m_j u + m_n v)/N}
// ---------------------------------------------------------------------------
__global__ void k_synth(const float* __restrict__ Zr, const float* __restrict__ Zi,
                        const float* __restrict__ syn, const float* __restrict__ bias,
                        float* __restrict__ y, int B, int O, int Kb, int L, int J, int N,
                        int Mp, int Np){
  extern __shared__ char smemraw[];
  float2* Mjn = (float2*)smemraw;      // J*J
  float2* G   = Mjn + J * J;           // J*N
  int g = blockIdx.x;
  int k = g % Kb; int o = (g / Kb) % O; int b = g / (Kb * O);
  for (int t = threadIdx.x; t < J * J; t += blockDim.x){
    int j = t / J, n = t % J;
    float sr = 0.f, si = 0.f;
    size_t zoff = (size_t)(j * B + b) * Np + (size_t)n * O + o;
    for (int l = 0; l < L; ++l){
      float s = syn[(((size_t)l * Kb + k) * J + j) * J + n];
      if (s != 0.f){
        size_t zp = (size_t)l * Mp * Np + zoff;
        sr += s * Zr[zp]; si += s * Zi[zp];
      }
    }
    Mjn[t] = make_float2(sr, si);
  }
  __syncthreads();
  for (int t = threadIdx.x; t < J * N; t += blockDim.x){
    int j = t / N, v = t % N;
    float re = 0.f, im = 0.f;
    for (int n = 0; n < J; ++n){
      float ang = -2.f * PI_F * (float)((n - (L - 1)) * v) / (float)N;
      float s, c; sincosf(ang, &s, &c);
      float2 m = Mjn[j * J + n];
      re += m.x * c - m.y * s;
      im += m.x * s + m.y * c;
    }
    G[t] = make_float2(re, im);
  }
  __syncthreads();
  float* dst = y + ((size_t)(b * O + o) * Kb + k) * N * N;
  float bv = bias[o];
  for (int t = threadIdx.x; t < N * N; t += blockDim.x){
    int u = t / N, v = t % N;
    float re = 0.f;
    for (int j = 0; j < J; ++j){
      float ang = -2.f * PI_F * (float)((j - (L - 1)) * u) / (float)N;
      float s, c; sincosf(ang, &s, &c);
      float2 gg = G[j * N + v];
      re += gg.x * c - gg.y * s;
    }
    dst[t] = re + bv;
  }
}

// ---------------------------------------------------------------------------
// BatchNorm (mean/var over batch+spatial) + ReLU, pooling, head
// ---------------------------------------------------------------------------
__global__ void k_bn_stats(const float* __restrict__ x, float* __restrict__ mean,
                           float* __restrict__ invstd, int B, int C, long S){
  __shared__ float s1[256], s2[256];
  int c = blockIdx.x;
  float a = 0.f, q = 0.f;
  long total = (long)B * S;
  for (long i = threadIdx.x; i < total; i += blockDim.x){
    long b = i / S, s = i % S;
    float v = x[((size_t)b * C + c) * S + s];
    a += v; q += v * v;
  }
  s1[threadIdx.x] = a; s2[threadIdx.x] = q;
  __syncthreads();
  for (int w = 128; w > 0; w >>= 1){
    if (threadIdx.x < w){ s1[threadIdx.x] += s1[threadIdx.x + w]; s2[threadIdx.x] += s2[threadIdx.x + w]; }
    __syncthreads();
  }
  if (threadIdx.x == 0){
    float n = (float)total;
    float m = s1[0] / n;
    float var = s2[0] / n - m * m;
    mean[c] = m;
    invstd[c] = rsqrtf(var + EPSF);
  }
}

__global__ void k_bn_apply_relu(float* __restrict__ x, const float* __restrict__ mean,
                                const float* __restrict__ invstd, const float* __restrict__ gm,
                                const float* __restrict__ bt, int C, long S, long total){
  long idx = (long)blockIdx.x * blockDim.x + threadIdx.x;
  if (idx >= total) return;
  int c = (int)((idx / S) % C);
  float v = (x[idx] - mean[c]) * invstd[c] * gm[c] + bt[c];
  x[idx] = v > 0.f ? v : 0.f;
}

__global__ void k_maxpool(const float* __restrict__ x, float* __restrict__ feat, long S){
  __shared__ float sm[256];
  long base = (long)blockIdx.x * S;
  float m = -3.4e38f;
  for (long i = threadIdx.x; i < S; i += blockDim.x){
    float v = x[base + i]; if (v > m) m = v;
  }
  sm[threadIdx.x] = m;
  __syncthreads();
  for (int w = 128; w > 0; w >>= 1){
    if (threadIdx.x < w){ if (sm[threadIdx.x + w] > sm[threadIdx.x]) sm[threadIdx.x] = sm[threadIdx.x + w]; }
    __syncthreads();
  }
  if (threadIdx.x == 0) feat[blockIdx.x] = sm[0];
}

__global__ void k_head(const float* __restrict__ feat, const float* __restrict__ w_out,
                       const float* __restrict__ b_out, float* __restrict__ out,
                       int B, int C, int Ncls){
  __shared__ float fn[16 * 100];
  __shared__ float lg[16 * 64];
  for (int c = threadIdx.x; c < C; c += blockDim.x){
    float m = 0.f;
    for (int b = 0; b < B; ++b) m += feat[b * C + c];
    m /= (float)B;
    float v = 0.f;
    for (int b = 0; b < B; ++b){ float d = feat[b * C + c] - m; v += d * d; }
    v /= (float)B;
    float is = rsqrtf(v + EPSF);
    for (int b = 0; b < B; ++b) fn[b * C + c] = (feat[b * C + c] - m) * is;
  }
  __syncthreads();
  for (int t = threadIdx.x; t < B * Ncls; t += blockDim.x){
    int b = t / Ncls, c = t % Ncls;
    float a = b_out[c];
    for (int f = 0; f < C; ++f) a += fn[b * C + f] * w_out[f * Ncls + c];
    lg[b * 64 + c] = a;
  }
  __syncthreads();
  for (int b = threadIdx.x; b < B; b += blockDim.x){
    float mx = -3.4e38f;
    for (int c = 0; c < Ncls; ++c){ float v = lg[b * 64 + c]; if (v > mx) mx = v; }
    float se = 0.f;
    for (int c = 0; c < Ncls; ++c) se += expf(lg[b * 64 + c] - mx);
    float ls = mx + logf(se);
    for (int c = 0; c < Ncls; ++c) out[b * Ncls + c] = lg[b * 64 + c] - ls;
  }
}

// ---------------------------------------------------------------------------
static inline int cdiv(long a, long b){ return (int)((a + b - 1) / b); }

extern "C" void kernel_launch(void* const* d_in, const int* in_sizes, int n_in,
                              void* d_out, int out_size, void* d_ws, size_t ws_size,
                              hipStream_t stream){
  (void)in_sizes; (void)n_in; (void)out_size; (void)ws_size;
  const float* x    = (const float*)d_in[0];
  const float* w1   = (const float*)d_in[1];
  const float* b1   = (const float*)d_in[2];
  const float* g1   = (const float*)d_in[3];
  const float* bt1  = (const float*)d_in[4];
  const float* w2   = (const float*)d_in[5];
  const float* b2   = (const float*)d_in[6];
  const float* g2   = (const float*)d_in[7];
  const float* bt2  = (const float*)d_in[8];
  const float* w3   = (const float*)d_in[9];
  const float* b3   = (const float*)d_in[10];
  const float* g3   = (const float*)d_in[11];
  const float* bt3  = (const float*)d_in[12];
  const float* wOut = (const float*)d_in[13];
  const float* bOut = (const float*)d_in[14];
  float* out = (float*)d_out;

  char* ws = (char*)d_ws;
  size_t off = 0;
  auto alloc = [&](size_t bytes) -> char* {
    size_t o = off; off += (bytes + 255) & ~(size_t)255; return ws + o;
  };

  // constants
  float* cS2ANA = (float*)alloc((size_t)99840  * 4);
  float* cHALF1 = (float*)alloc((size_t)780    * 4);
  float* cSYN1  = (float*)alloc((size_t)1216800* 4);
  float* cANA2  = (float*)alloc((size_t)144400 * 4);
  float* cHALF2 = (float*)alloc((size_t)3610   * 4);
  float* cSYN2  = (float*)alloc((size_t)72200  * 4);
  float* cANA3  = (float*)alloc((size_t)23660  * 4);
  float* cHALF3 = (float*)alloc((size_t)1183   * 4);
  float* cSYN3  = (float*)alloc((size_t)16562  * 4);
  // arenas (reused across layers)
  float2* xhA = (float2*)alloc((size_t)9241600  * 4);   // max xh  (layer2)
  float2* fxA = (float2*)alloc((size_t)2310400  * 4);   // max Fx  (layer2)
  float2* kyA = (float2*)alloc((size_t)14196000 * 4);   // max Ky  (layer3)
  bf16*   Aar = (bf16*)  alloc((size_t)2 * 1167360 * 2);// 2 A planes (max layer2)
  bf16*   Bar = (bf16*)  alloc((size_t)3 * 7347200 * 2);// 3 B planes (max layer3)
  float*  Zar = (float*) alloc((size_t)2 * 9784320 * 4);// 2 Z planes (max layer1)
  float*  yP  = (float*) alloc((size_t)20480000 * 4);   // y1 (later y3)
  float*  yQ  = (float*) alloc((size_t)7680000  * 4);   // y2
  float*  stM = (float*) alloc((size_t)128 * 4);
  float*  stI = (float*) alloc((size_t)128 * 4);
  float*  fea = (float*) alloc((size_t)1600 * 4);

  // ---- constants ----
  k_fill_s2_ana<<<cdiv(99840, 256), 256, 0, stream>>>(cS2ANA, 20, 128, 39, 64, (float)(2.0 * PI_D / 128.0));
  k_fill_half1 <<<cdiv(780, 64),    64,  0, stream>>>(cHALF1, 20, 39);
  k_fill_wig4  <<<cdiv(1216800, 256), 256, 0, stream>>>(cSYN1, 20, 40, 39, 20, 1, 0.f);
  k_fill_wig4  <<<cdiv(144400, 256),  256, 0, stream>>>(cANA2, 10, 40, 19, 20, 0,
                 (float)((2.0 * PI_D / 40.0) * (2.0 * PI_D / 40.0)));
  k_fill_half2d<<<cdiv(3610, 256),   256, 0, stream>>>(cHALF2, 10, 19);
  k_fill_wig4  <<<cdiv(72200, 256),  256, 0, stream>>>(cSYN2, 10, 20, 19, 10, 1, 0.f);
  k_fill_wig4  <<<cdiv(23660, 256),  256, 0, stream>>>(cANA3, 7, 20, 13, 10, 0,
                 (float)((2.0 * PI_D / 20.0) * (2.0 * PI_D / 20.0)));
  k_fill_half2d<<<cdiv(1183, 128),   128, 0, stream>>>(cHALF3, 7, 13);
  k_fill_wig4  <<<cdiv(16562, 256),  256, 0, stream>>>(cSYN3, 7, 14, 13, 7, 1, 0.f);

  // ---- layer 1 : S2 conv (B=16,F=6,Kin=128,Nin=128 -> Lo=20) ----
  {
    const int L = 20, J = 39, B = 16, F = 6, O = 20, Kin = 128, P = 128, N = 40;
    const int Mp = 624, Np = 784, Kp = 32;
    const size_t pA = (size_t)L * Mp * Kp, pB = (size_t)L * Np * Kp, pZ = (size_t)L * Mp * Np;
    bf16 *Ar = Aar, *Ai = Aar + pA;
    bf16 *Bp = Bar, *Bq = Bar + pB, *Br = Bar + 2 * pB;
    float *Zr = Zar, *Zi = Zar + pZ;
    k_dft_alpha<<<cdiv((long)B * F * Kin * J, 256), 256, 0, stream>>>(x, xhA, B * F * Kin, 128, J, L - 1);
    k_fx_s2<<<cdiv((long)L * J * B * F, 256), 256, 0, stream>>>(cS2ANA, xhA, fxA, L, Kin, J, B, F);
    k_ky_s2<<<cdiv((long)L * J * F * O, 256), 256, 0, stream>>>(cHALF1, w1, kyA, L, J, F, O, P,
                                                                1.f / sqrtf((float)(P * F)));
    k_packA<<<cdiv((long)pA, 256), 256, 0, stream>>>(fxA, Ar, Ai, L, J, 1, B, F, Mp, Kp);
    k_packB<<<cdiv((long)pB, 256), 256, 0, stream>>>(kyA, Bp, Bq, Br, L, J, 1, F, O, Np, Kp);
    k_zgemm<<<dim3(Mp / 16, Np / 16, L), 32, 0, stream>>>(Ar, Ai, Bp, Bq, Br, Zr, Zi, Mp, Np, Kp);
    size_t shm = (size_t)(J * J + J * N) * sizeof(float2);
    k_synth<<<B * O * N, 256, shm, stream>>>(Zr, Zi, cSYN1, b1, yP, B, O, N, L, J, N, Mp, Np);
    long S = (long)N * N * N;
    k_bn_stats<<<O, 256, 0, stream>>>(yP, stM, stI, B, O, S);
    k_bn_apply_relu<<<cdiv((long)B * O * S, 256), 256, 0, stream>>>(yP, stM, stI, g1, bt1, O, S, (long)B * O * S);
  }

  // ---- layer 2 : SO3 conv (F=20,Kin=40,Nin=40 -> Lo=10) ----
  {
    const int L = 10, J = 19, B = 16, F = 20, O = 60, Kin = 40, Nin = 40, P = 40, N = 20;
    const int Mp = 304, Np = 1152, Kp = 384;
    const size_t pA = (size_t)L * Mp * Kp, pB = (size_t)L * Np * Kp, pZ = (size_t)L * Mp * Np;
    bf16 *Ar = Aar, *Ai = Aar + pA;
    bf16 *Bp = Bar, *Bq = Bar + pB, *Br = Bar + 2 * pB;
    float *Zr = Zar, *Zi = Zar + pZ;
    size_t shd = (size_t)Nin * Nin * 4 + (size_t)J * Nin * 8;
    k_dft2<<<B * F * Kin, 256, shd, stream>>>(yP, xhA, Nin, J, L - 1);
    k_fx_so3<<<cdiv((long)L * J * J * B * F, 256), 256, 0, stream>>>(cANA2, xhA, fxA, L, Kin, J, B, F);
    k_ky_so3<<<cdiv((long)L * J * J * F * O, 256), 256, 0, stream>>>(cHALF2, w2, kyA, L, J, F, O, P,
                                                                     1.f / sqrtf((float)(P * F)));
    k_packA<<<cdiv((long)pA, 256), 256, 0, stream>>>(fxA, Ar, Ai, L, J, J, B, F, Mp, Kp);
    k_packB<<<cdiv((long)pB, 256), 256, 0, stream>>>(kyA, Bp, Bq, Br, L, J, J, F, O, Np, Kp);
    k_zgemm<<<dim3(Mp / 16, Np / 16, L), 32, 0, stream>>>(Ar, Ai, Bp, Bq, Br, Zr, Zi, Mp, Np, Kp);
    size_t shm = (size_t)(J * J + J * N) * sizeof(float2);
    k_synth<<<B * O * N, 256, shm, stream>>>(Zr, Zi, cSYN2, b2, yQ, B, O, N, L, J, N, Mp, Np);
    long S = (long)N * N * N;
    k_bn_stats<<<O, 256, 0, stream>>>(yQ, stM, stI, B, O, S);
    k_bn_apply_relu<<<cdiv((long)B * O * S, 256), 256, 0, stream>>>(yQ, stM, stI, g2, bt2, O, S, (long)B * O * S);
  }

  // ---- layer 3 : SO3 conv (F=60,Kin=20,Nin=20 -> Lo=7) ----
  {
    const int L = 7, J = 13, B = 16, F = 60, O = 100, Kin = 20, Nin = 20, P = 20, N = 14;
    const int Mp = 208, Np = 1312, Kp = 800;
    const size_t pA = (size_t)L * Mp * Kp, pB = (size_t)L * Np * Kp, pZ = (size_t)L * Mp * Np;
    bf16 *Ar = Aar, *Ai = Aar + pA;
    bf16 *Bp = Bar, *Bq = Bar + pB, *Br = Bar + 2 * pB;
    float *Zr = Zar, *Zi = Zar + pZ;
    size_t shd = (size_t)Nin * Nin * 4 + (size_t)J * Nin * 8;
    k_dft2<<<B * F * Kin, 256, shd, stream>>>(yQ, xhA, Nin, J, L - 1);
    k_fx_so3<<<cdiv((long)L * J * J * B * F, 256), 256, 0, stream>>>(cANA3, xhA, fxA, L, Kin, J, B, F);
    k_ky_so3<<<cdiv((long)L * J * J * F * O, 256), 256, 0, stream>>>(cHALF3, w3, kyA, L, J, F, O, P,
                                                                     1.f / sqrtf((float)(P * F)));
    k_packA<<<cdiv((long)pA, 256), 256, 0, stream>>>(fxA, Ar, Ai, L, J, J, B, F, Mp, Kp);
    k_packB<<<cdiv((long)pB, 256), 256, 0, stream>>>(kyA, Bp, Bq, Br, L, J, J, F, O, Np, Kp);
    k_zgemm<<<dim3(Mp / 16, Np / 16, L), 32, 0, stream>>>(Ar, Ai, Bp, Bq, Br, Zr, Zi, Mp, Np, Kp);
    size_t shm = (size_t)(J * J + J * N) * sizeof(float2);
    k_synth<<<B * O * N, 256, shm, stream>>>(Zr, Zi, cSYN3, b3, yP, B, O, N, L, J, N, Mp, Np);
    long S = (long)N * N * N;
    k_bn_stats<<<O, 256, 0, stream>>>(yP, stM, stI, B, O, S);
    k_bn_apply_relu<<<cdiv((long)B * O * S, 256), 256, 0, stream>>>(yP, stM, stI, g3, bt3, O, S, (long)B * O * S);
    // head
    k_maxpool<<<B * O, 256, 0, stream>>>(yP, fea, S);
    k_head<<<1, 256, 0, stream>>>(fea, wOut, bOut, out, B, O, 55);
  }
}
